// additiveAttention_64922725647030
// MI455X (gfx1250) — matrix operand order repaired and checked by reading, not compiled, not run on hardware
//
#include <hip/hip_runtime.h>
#include <hip/hip_bf16.h>
#include <stdint.h>

// Problem constants (match reference)
#define NB   2048     // B
#define NT   200      // T
#define NTP  208      // T padded to multiple of 16
#define ND   256      // D
#define NK2  512      // 2*D (W1 row length)

typedef __attribute__((ext_vector_type(16))) __bf16 v16bf;
typedef __attribute__((ext_vector_type(8)))  float  v8f;

union FragBF16 { uint32_t u[8]; uint4 q[2]; v16bf v; };

// round-to-nearest-even f32 -> bf16, packed pair into u32
__device__ __forceinline__ uint32_t pack_bf16x2(float lo, float hi) {
    uint32_t a = __float_as_uint(lo);
    uint32_t b = __float_as_uint(hi);
    a = (a + 0x7FFFu + ((a >> 16) & 1u)) >> 16;
    b = (b + 0x7FFFu + ((b >> 16) & 1u)) >> 16;
    return (a & 0xFFFFu) | (b << 16);
}

// branch-free tanh: 1 - 2/(exp(2x)+1); exact saturation at +/-inf, no cmp/branch
__device__ __forceinline__ float fast_tanh(float x) {
    const float e = __expf(2.0f * x);                  // v_mul + v_exp_f32
    const float r = __builtin_amdgcn_rcpf(e + 1.0f);   // v_add + v_rcp_f32
    return __builtin_fmaf(-2.0f, r, 1.0f);             // v_fma
}

__device__ __forceinline__ float wave_reduce16(float v) {
    // sum across the 16-lane half-groups (xor 1,2,4,8 stays inside each half)
    v += __shfl_xor(v, 1, 32);
    v += __shfl_xor(v, 2, 32);
    v += __shfl_xor(v, 4, 32);
    v += __shfl_xor(v, 8, 32);
    return v;
}

__global__ __launch_bounds__(256, 1)
void din_attn_kernel(const float* __restrict__ query,        // [B, 1, D]
                     const float* __restrict__ ub,           // [B, T, D]
                     const unsigned char* __restrict__ mask, // [B, T] (bool)
                     const float* __restrict__ W1,           // [D, 2D] row-major
                     const float* __restrict__ W2,           // [1, D]
                     float* __restrict__ out)                // [B, D]
{
    __shared__ uint32_t lds_ub[NTP * (ND / 2)]; // bf16x2 packed, 208*128*4 = 104 KB
    __shared__ float    lds_q[ND];
    __shared__ float    lds_qp[ND];
    __shared__ float    lds_w2[ND];
    __shared__ float    lds_score[NTP];
    __shared__ float    lds_red[1];

    const int tid  = threadIdx.x;
    const int lane = tid & 31;
    const int wv   = tid >> 5;          // wave id 0..7
    const int ln   = lane & 15;
    const int b    = blockIdx.x;

    const float* ubb = ub + (size_t)b * NT * ND;

    // ---------------- Phase 0: stage q, W2, ub(bf16) into LDS -------------
    lds_q[tid]  = query[b * ND + tid];
    lds_w2[tid] = W2[tid];
    if (tid < NTP) lds_score[tid] = 0.0f;

    for (int i = tid; i < NT * (ND / 2); i += 256) {
        const int row = i >> 7;       // / 128
        const int c2  = i & 127;
        const float2 f = *reinterpret_cast<const float2*>(ubb + row * ND + c2 * 2);
        lds_ub[row * (ND / 2) + c2] = pack_bf16x2(f.x, f.y);
    }
    for (int i = NT * (ND / 2) + tid; i < NTP * (ND / 2); i += 256)
        lds_ub[i] = 0u;   // zero-pad rows 200..207

    __syncthreads();

    // ---------------- Phase 0b: qp[d] = sum_k q[k] * W1[d, k], k < 256 ----
    {
        float acc = 0.0f;
        const float4* wrow = reinterpret_cast<const float4*>(W1 + (size_t)tid * NK2);
        #pragma unroll 4
        for (int k4 = 0; k4 < ND / 4; ++k4) {
            const float4 w = wrow[k4];
            acc += lds_q[k4 * 4 + 0] * w.x + lds_q[k4 * 4 + 1] * w.y +
                   lds_q[k4 * 4 + 2] * w.z + lds_q[k4 * 4 + 3] * w.w;
        }
        lds_qp[tid] = acc;
    }
    __syncthreads();

    // ---------------- Preload per-wave W1u B-fragments into VGPRs ---------
    // Wave wv owns output columns [wv*32, wv*32+32): two 16-wide N-tiles.
    // B layout (32x16 bf16): lane holds column n = lane&15;
    //   VGPR 0-3: k = khalf + 2v,2v+1 (khalf = lane<16?0:8); VGPR 4-7: +16 (matches A).
    const int nb0   = wv * 32;
    const int nb1   = nb0 + 16;
    const int khalf = (lane < 16) ? 0 : 8;

    FragBF16 Bf[2][8];
    #pragma unroll
    for (int nt = 0; nt < 2; ++nt) {
        const int n = (nt ? nb1 : nb0) + ln;
        const float* wrow = W1 + (size_t)n * NK2 + ND;   // W1u row n (cols 256..511)
        #pragma unroll
        for (int kk = 0; kk < 8; ++kk) {
            #pragma unroll
            for (int v = 0; v < 8; ++v) {
                const int k = kk * 32 + khalf + 2 * v + ((v >= 4) ? 8 : 0);
                const float2 f = *reinterpret_cast<const float2*>(wrow + k);
                Bf[nt][kk].u[v] = pack_bf16x2(f.x, f.y);
            }
        }
    }

    const float qp0 = lds_qp[nb0 + ln];
    const float qp1 = lds_qp[nb1 + ln];
    const float w20 = lds_w2[nb0 + ln];
    const float w21 = lds_w2[nb1 + ln];

    // ---------------- Phase 1: ubp = ub * W1u^T + qp, tanh, dot W2 --------
    for (int mt = 0; mt < NTP / 16; ++mt) {
        const int mbase = mt * 16;
        const int trow  = mbase + ln;    // A-matrix row (t index) for this lane

        // A layout (16x32 bf16): lane holds row m = lane&15;
        //   VGPR 0-3: k = (lane<16?0:8) + 2v,2v+1 ; VGPR 4-7: +16.
        // As u32 indices per row: [kk*16 + half4 .. +3] and [kk*16+8+half4 .. +3]
        const uint4* ldsrow = reinterpret_cast<const uint4*>(
            lds_ub + trow * (ND / 2) + ((lane < 16) ? 0 : 4));

        v8f c0, c1;
        #pragma unroll
        for (int r = 0; r < 8; ++r) { c0[r] = qp0; c1[r] = qp1; }

        // double-buffered A fragments so DS loads overlap the tensor pipe
        FragBF16 Abuf[2];
        Abuf[0].q[0] = ldsrow[0];
        Abuf[0].q[1] = ldsrow[2];
        #pragma unroll
        for (int kk = 0; kk < 8; ++kk) {
            if (kk < 7) {
                Abuf[(kk + 1) & 1].q[0] = ldsrow[(kk + 1) * 4];
                Abuf[(kk + 1) & 1].q[1] = ldsrow[(kk + 1) * 4 + 2];
            }
            const v16bf av = Abuf[kk & 1].v;
            c0 = __builtin_amdgcn_wmma_f32_16x16x32_bf16(
                     false, av, false, Bf[0][kk].v, (short)0, c0, false, false);
            c1 = __builtin_amdgcn_wmma_f32_16x16x32_bf16(
                     false, av, false, Bf[1][kk].v, (short)0, c1, false, false);
        }

        // score[t] += sum_n tanh(h[t,n]) * W2[n] over this wave's 32 columns.
        // C/D layout: VGPR r -> row m=r (lanes 0-15), m=r+8 (lanes 16-31); N = lane&15.
        #pragma unroll
        for (int r = 0; r < 8; ++r) {
            float vsum = fast_tanh(c0[r]) * w20 + fast_tanh(c1[r]) * w21;
            vsum = wave_reduce16(vsum);
            if (ln == 0)
                atomicAdd(&lds_score[mbase + r + ((lane >> 4) << 3)], vsum);
        }
    }
    __syncthreads();

    // ---------------- Phase 2: mask (->0) then softmax over t < 200 -------
    if (tid < NT) {
        float s = lds_score[tid];
        if (mask[(size_t)b * NT + tid]) s = 0.0f;   // masked_fill BEFORE softmax
        lds_score[tid] = s;
    }
    __syncthreads();

    if (wv == 0) {
        float m = -3.0e38f;
        for (int t = lane; t < NT; t += 32) m = fmaxf(m, lds_score[t]);
        m = fmaxf(m, __shfl_xor(m, 1, 32));
        m = fmaxf(m, __shfl_xor(m, 2, 32));
        m = fmaxf(m, __shfl_xor(m, 4, 32));
        m = fmaxf(m, __shfl_xor(m, 8, 32));
        m = fmaxf(m, __shfl_xor(m, 16, 32));

        float s = 0.0f;
        for (int t = lane; t < NT; t += 32) {
            const float e = __expf(lds_score[t] - m);
            lds_score[t] = e;
            s += e;
        }
        s += __shfl_xor(s, 1, 32);
        s += __shfl_xor(s, 2, 32);
        s += __shfl_xor(s, 4, 32);
        s += __shfl_xor(s, 8, 32);
        s += __shfl_xor(s, 16, 32);
        if (lane == 0) lds_red[0] = s;
    }
    __syncthreads();
    const float inv = 1.0f / lds_red[0];

    // ---------------- Phase 3: out[d] = sum_t attn[t] * ub[t, d] ----------
    {
        float acc = 0.0f;
        const float* col = ubb + tid;          // thread = column d, coalesced
        #pragma unroll 4
        for (int t = 0; t < NT; ++t)
            acc += lds_score[t] * col[t * ND];
        out[(size_t)b * ND + tid] = acc * inv;
    }
}

extern "C" void kernel_launch(void* const* d_in, const int* in_sizes, int n_in,
                              void* d_out, int out_size, void* d_ws, size_t ws_size,
                              hipStream_t stream) {
    const float*         query = (const float*)d_in[0];         // [B,1,D]
    const float*         ub    = (const float*)d_in[1];         // [B,T,D]
    const unsigned char* mask  = (const unsigned char*)d_in[2]; // [B,T] bool
    const float*         W1    = (const float*)d_in[3];         // [D,2D]
    const float*         W2    = (const float*)d_in[4];         // [1,D]
    float*               out   = (float*)d_out;                 // [B,D]

    din_attn_kernel<<<NB, 256, 0, stream>>>(query, ub, mask, W1, W2, out);
}
